// SpatialProcessor_3530463117993
// MI455X (gfx1250) — matrix operand order repaired and checked by reading, not compiled, not run on hardware
//
#include <hip/hip_runtime.h>
#include <hip/hip_bf16.h>

typedef __attribute__((ext_vector_type(16))) _Float16 v16h;
typedef __attribute__((ext_vector_type(8)))  _Float16 v8h;
typedef __attribute__((ext_vector_type(8)))  float    v8f;

#define B_SZ  4
#define N_SZ  2048
#define FIN   32
#define DOUT  64
#define H1    4
#define D1    32      // layer-1 head dim
#define C1    128     // 2*D = layer-1 output width
#define D2    64      // layer-2 head dim (1 head)

// ---- 16-lane (half-wave) reductions: xor masks <16 keep lanes 0-15 / 16-31 separate,
// matching the WMMA C-layout row striping (row R lives on one 16-lane half). ----
static __device__ __forceinline__ float red_max16(float v) {
#pragma unroll
  for (int m = 8; m >= 1; m >>= 1) v = fmaxf(v, __shfl_xor(v, m, 32));
  return v;
}
static __device__ __forceinline__ float red_sum16(float v) {
#pragma unroll
  for (int m = 8; m >= 1; m >>= 1) v += __shfl_xor(v, m, 32);
  return v;
}

static __device__ __forceinline__ v8f wmma_f16(v16h a, v16h b, v8f c) {
  // D = A(16x32 f16) x B(32x16 f16) + C(16x16 f32)
  return __builtin_amdgcn_wmma_f32_16x16x32_f16(false, a, false, b, (short)0, c,
                                                false, false);
}

// =====================================================================
// Kernel A: layer-1 Q/K/V projections + residual projection.
// One block = 16-row strip of x (f32 [B*N,32] -> LDS f16). K=FIN=32 fits one
// WMMA exactly. 28 output col-tiles (8 q + 8 k + 8 v + 4 proj) spread over 8
// waves; the tile index is wave-uniform, so it is forced through
// readfirstlane to get scalar branches instead of EXEC-masked ones.
// V is written transposed ([B][128][N]) so attention's P x V B-operand loads
// are contiguous along j.
// =====================================================================
__global__ void __launch_bounds__(256)
qkv1_proj_kernel(const float* __restrict__ x,
                 const float* __restrict__ Wq, const float* __restrict__ bq,
                 const float* __restrict__ Wk, const float* __restrict__ bk,
                 const float* __restrict__ Wv, const float* __restrict__ bv,
                 const float* __restrict__ Wp, const float* __restrict__ bp,
                 _Float16* __restrict__ q1, _Float16* __restrict__ k1,
                 _Float16* __restrict__ v1T, float* __restrict__ proj) {
  __shared__ _Float16 xs[16 * FIN];
  const int strip0 = blockIdx.x * 16;
  const int tid = threadIdx.x;
  {
    const int e = tid * 2;
    xs[e]     = (_Float16)x[(size_t)strip0 * FIN + e];
    xs[e + 1] = (_Float16)x[(size_t)strip0 * FIN + e + 1];
  }
  __syncthreads();
  const int lane = tid & 31;
  const int wave = __builtin_amdgcn_readfirstlane(tid >> 5);  // wave-uniform
  const int hl = lane >> 4, lr = lane & 15;

  // A fragment (ISA 16-bit A layout): lanes<16 row=lane,K={0..7,16..23};
  // lanes>=16 K={8..15,24..31}
  v16h a;
  {
    const int ko = hl ? 8 : 0;
#pragma unroll
    for (int i = 0; i < 8; ++i) {
      a[i]     = xs[lr * FIN + ko + i];
      a[i + 8] = xs[lr * FIN + ko + 16 + i];
    }
  }

  for (int t = wave; t < 28; t += 8) {
    const int tu = __builtin_amdgcn_readfirstlane(t);
    const float *W, *bias;
    int col0, mode;
    if (tu < 8)       { W = Wq; bias = bq; col0 = tu * 16;        mode = 0; }
    else if (tu < 16) { W = Wk; bias = bk; col0 = (tu - 8) * 16;  mode = 1; }
    else if (tu < 24) { W = Wv; bias = bv; col0 = (tu - 16) * 16; mode = 2; }
    else              { W = Wp; bias = bp; col0 = (tu - 24) * 16; mode = 3; }
    const int ldw  = (mode == 3) ? DOUT : C1;
    const int coln = col0 + lr;
    // B fragment: lane holds col N=coln, K={0..7,16..23} or {8..15,24..31}
    v16h bf;
    const int kb = hl ? 8 : 0;
#pragma unroll
    for (int kk = 0; kk < 16; ++kk)
      bf[kk] = (_Float16)W[(kb + (kk & 7) + (kk >> 3) * 16) * ldw + coln];
    v8f c = {};
    c = wmma_f16(a, bf, c);
    const float bb = bias[coln];
    if (mode <= 1) {               // q1 / k1, row-major f16
      _Float16* dst = (mode == 0) ? q1 : k1;
#pragma unroll
      for (int r = 0; r < 8; ++r) {
        const int row = strip0 + r + 8 * hl;
        dst[(size_t)row * C1 + coln] = (_Float16)(c[r] + bb);
      }
    } else if (mode == 2) {        // v transposed: [B][128][N]
#pragma unroll
      for (int r = 0; r < 8; ++r) {
        const int row = strip0 + r + 8 * hl;
        const int bi = row / N_SZ, nn = row % N_SZ;
        v1T[((size_t)bi * C1 + coln) * N_SZ + nn] = (_Float16)(c[r] + bb);
      }
    } else {                       // residual projection, f32
#pragma unroll
      for (int r = 0; r < 8; ++r) {
        const int row = strip0 + r + 8 * hl;
        proj[(size_t)row * DOUT + coln] = c[r] + bb;
      }
    }
  }
}

// =====================================================================
// Kernel B: layer-1 fused attention (flash-style online softmax) + ReLU.
// grid = (N/128, H1, B); 8 waves/block, each wave owns one 16-row i-tile.
// Per 32-j chunk: 2 score WMMAs (K=32=head dim), softmax update, P staged
// through LDS (C-layout -> A-layout), 2 accumulating P x V WMMAs.
// Scores never touch memory; adj (16MB) is L2-resident (192MB L2).
// =====================================================================
__global__ void __launch_bounds__(256)
attn1_kernel(const _Float16* __restrict__ q1, const _Float16* __restrict__ k1,
             const _Float16* __restrict__ v1T, const float* __restrict__ adj,
             _Float16* __restrict__ h1) {
  __shared__ _Float16 Pl[8][16 * 32];
  const int lane = threadIdx.x & 31, wave = threadIdx.x >> 5;
  const int hl = lane >> 4, lr = lane & 15;
  const int i0 = (blockIdx.x * 8 + wave) * 16;
  const int h  = blockIdx.y;
  const int b  = blockIdx.z;
  const float scale = 0.17677669529663687f;  // 1/sqrt(32)

  // Q fragment (A layout), loaded once
  v16h qf;
  {
    const _Float16* qp = q1 + ((size_t)(b * N_SZ + i0 + lr) * C1 + h * D1);
    const int ko = hl ? 16 : 0;
    v8h lo = *(const v8h*)(qp + ko);
    v8h hi = *(const v8h*)(qp + ko + 8);
#pragma unroll
    for (int i = 0; i < 8; ++i) { qf[i] = lo[i]; qf[i + 8] = hi[i]; }
  }

  float m[8], l[8];
  v8f acc0 = {}, acc1 = {};
#pragma unroll
  for (int r = 0; r < 8; ++r) { m[r] = -1e30f; l[r] = 0.f; }
  _Float16* P = &Pl[wave][0];

  for (int j0 = 0; j0 < N_SZ; j0 += 32) {
    v8f s[2];
#pragma unroll
    for (int half = 0; half < 2; ++half) {
      const int jt = j0 + half * 16;
      const int kk = hl ? 16 : 0;
      // K^T B fragment: lane holds col j, K contiguous along head dim
      v16h kf = *(const v16h*)(k1 + ((size_t)(b * N_SZ + jt + lr) * C1 + h * D1 + kk));
      v8f sc = {};
      sc = wmma_f16(qf, kf, sc);
#pragma unroll
      for (int r = 0; r < 8; ++r) {
        const int i = i0 + r + 8 * hl;
        const float av = adj[(size_t)i * N_SZ + jt + lr];
        sc[r] = sc[r] * scale + (av == 0.f ? -1e9f : 0.f);
      }
      s[half] = sc;
    }
    // online softmax update (per C-layout row r)
#pragma unroll
    for (int r = 0; r < 8; ++r) {
      const float mx    = red_max16(fmaxf(s[0][r], s[1][r]));
      const float mnew  = fmaxf(m[r], mx);
      const float alpha = __expf(m[r] - mnew);
      const float p0 = __expf(s[0][r] - mnew);
      const float p1 = __expf(s[1][r] - mnew);
      l[r] = l[r] * alpha + red_sum16(p0 + p1);
      m[r] = mnew;
      acc0[r] *= alpha;
      acc1[r] *= alpha;
      const int prow = r + 8 * hl;
      P[prow * 32 + lr]      = (_Float16)p0;
      P[prow * 32 + 16 + lr] = (_Float16)p1;
    }
    asm volatile("s_wait_dscnt 0" ::: "memory");
    // P as A fragment
    v16h pf;
    {
      const int ko = hl ? 16 : 0;
#pragma unroll
      for (int i = 0; i < 8; ++i) {
        pf[i]     = P[lr * 32 + ko + i];
        pf[i + 8] = P[lr * 32 + ko + 8 + i];
      }
    }
    // V B fragments from transposed V (contiguous along j)
    const int jr = j0 + (hl ? 16 : 0);
    v16h v0 = *(const v16h*)(v1T + ((size_t)(b * C1 + h * D1 + lr) * N_SZ + jr));
    v16h v1 = *(const v16h*)(v1T + ((size_t)(b * C1 + h * D1 + 16 + lr) * N_SZ + jr));
    acc0 = wmma_f16(pf, v0, acc0);
    acc1 = wmma_f16(pf, v1, acc1);
  }
  // normalize + ReLU, emit f16 for layer-2 projections
#pragma unroll
  for (int r = 0; r < 8; ++r) {
    const int i = i0 + r + 8 * hl;
    const float inv = 1.f / (l[r] + 1e-20f);
    h1[(size_t)(b * N_SZ + i) * C1 + h * D1 + lr]      = (_Float16)fmaxf(acc0[r] * inv, 0.f);
    h1[(size_t)(b * N_SZ + i) * C1 + h * D1 + 16 + lr] = (_Float16)fmaxf(acc1[r] * inv, 0.f);
  }
}

// =====================================================================
// Kernel C: layer-2 Q/K/V projections. K=128 -> 4 accumulating WMMAs.
// 12 col-tiles over 8 waves (uniform via readfirstlane). V stored transposed.
// =====================================================================
__global__ void __launch_bounds__(256)
qkv2_kernel(const _Float16* __restrict__ h1,
            const float* __restrict__ Wq, const float* __restrict__ bq,
            const float* __restrict__ Wk, const float* __restrict__ bk,
            const float* __restrict__ Wv, const float* __restrict__ bv,
            _Float16* __restrict__ q2, _Float16* __restrict__ k2,
            _Float16* __restrict__ v2T) {
  __shared__ _Float16 hs[16 * C1];
  const int strip0 = blockIdx.x * 16;
  const int tid = threadIdx.x;
  {
    const int e = tid * 8;  // 2048 halves / 256 threads
    *(v8h*)(hs + e) = *(const v8h*)(h1 + (size_t)strip0 * C1 + e);
  }
  __syncthreads();
  const int lane = tid & 31;
  const int wave = __builtin_amdgcn_readfirstlane(tid >> 5);
  const int hl = lane >> 4, lr = lane & 15;

  v16h a[4];
  {
    const int ko = hl ? 8 : 0;
#pragma unroll
    for (int c = 0; c < 4; ++c)
#pragma unroll
      for (int i = 0; i < 8; ++i) {
        a[c][i]     = hs[lr * C1 + c * 32 + ko + i];
        a[c][i + 8] = hs[lr * C1 + c * 32 + ko + 16 + i];
      }
  }
  for (int t = wave; t < 12; t += 8) {
    const int tu = __builtin_amdgcn_readfirstlane(t);
    const float *W, *bias;
    int col0, mode;
    if (tu < 4)      { W = Wq; bias = bq; col0 = tu * 16;       mode = 0; }
    else if (tu < 8) { W = Wk; bias = bk; col0 = (tu - 4) * 16; mode = 1; }
    else             { W = Wv; bias = bv; col0 = (tu - 8) * 16; mode = 2; }
    const int coln = col0 + lr;
    v8f c = {};
#pragma unroll
    for (int kc = 0; kc < 4; ++kc) {
      v16h bf;
      const int kb = kc * 32 + (hl ? 8 : 0);
#pragma unroll
      for (int kk = 0; kk < 16; ++kk)
        bf[kk] = (_Float16)W[(kb + (kk & 7) + (kk >> 3) * 16) * D2 + coln];
      c = wmma_f16(a[kc], bf, c);
    }
    const float bb = bias[coln];
    if (mode <= 1) {
      _Float16* dst = (mode == 0) ? q2 : k2;
#pragma unroll
      for (int r = 0; r < 8; ++r) {
        const int row = strip0 + r + 8 * hl;
        dst[(size_t)row * D2 + coln] = (_Float16)(c[r] + bb);
      }
    } else {
#pragma unroll
      for (int r = 0; r < 8; ++r) {
        const int row = strip0 + r + 8 * hl;
        const int bi = row / N_SZ, nn = row % N_SZ;
        v2T[((size_t)bi * D2 + coln) * N_SZ + nn] = (_Float16)(c[r] + bb);
      }
    }
  }
}

// =====================================================================
// Kernel D: layer-2 fused attention (1 head, d=64) + residual + LayerNorm.
// Score: 2 accumulating WMMAs (K=64); output: 4 d-tiles of P x V WMMAs.
// LayerNorm over 64 cols = per-row 16-lane reduction across 4 tile values.
// =====================================================================
__global__ void __launch_bounds__(256)
attn2_ln_kernel(const _Float16* __restrict__ q2, const _Float16* __restrict__ k2,
                const _Float16* __restrict__ v2T, const float* __restrict__ adj,
                const float* __restrict__ proj,
                const float* __restrict__ gamma, const float* __restrict__ beta,
                float* __restrict__ out) {
  __shared__ _Float16 Pl[8][16 * 32];
  const int lane = threadIdx.x & 31, wave = threadIdx.x >> 5;
  const int hl = lane >> 4, lr = lane & 15;
  const int i0 = (blockIdx.x * 8 + wave) * 16;
  const int b  = blockIdx.z;
  const float scale = 0.125f;  // 1/sqrt(64)

  v16h qa[2];
  {
    const _Float16* qp = q2 + (size_t)(b * N_SZ + i0 + lr) * D2;
    const int ko = hl ? 16 : 0;
#pragma unroll
    for (int c = 0; c < 2; ++c) {
      v8h lo = *(const v8h*)(qp + c * 32 + ko);
      v8h hi = *(const v8h*)(qp + c * 32 + ko + 8);
#pragma unroll
      for (int i = 0; i < 8; ++i) { qa[c][i] = lo[i]; qa[c][i + 8] = hi[i]; }
    }
  }

  float m[8], l[8];
  v8f zero = {};
  v8f acc[4];
#pragma unroll
  for (int t = 0; t < 4; ++t) acc[t] = zero;
#pragma unroll
  for (int r = 0; r < 8; ++r) { m[r] = -1e30f; l[r] = 0.f; }
  _Float16* P = &Pl[wave][0];

  for (int j0 = 0; j0 < N_SZ; j0 += 32) {
    v8f s[2];
#pragma unroll
    for (int half = 0; half < 2; ++half) {
      const int jt = j0 + half * 16;
      const int kk = hl ? 16 : 0;
      const _Float16* kp = k2 + (size_t)(b * N_SZ + jt + lr) * D2;
      v16h k0 = *(const v16h*)(kp + kk);
      v16h k1f = *(const v16h*)(kp + 32 + kk);
      v8f sc = {};
      sc = wmma_f16(qa[0], k0, sc);
      sc = wmma_f16(qa[1], k1f, sc);
#pragma unroll
      for (int r = 0; r < 8; ++r) {
        const int i = i0 + r + 8 * hl;
        const float av = adj[(size_t)i * N_SZ + jt + lr];
        sc[r] = sc[r] * scale + (av == 0.f ? -1e9f : 0.f);
      }
      s[half] = sc;
    }
#pragma unroll
    for (int r = 0; r < 8; ++r) {
      const float mx    = red_max16(fmaxf(s[0][r], s[1][r]));
      const float mnew  = fmaxf(m[r], mx);
      const float alpha = __expf(m[r] - mnew);
      const float p0 = __expf(s[0][r] - mnew);
      const float p1 = __expf(s[1][r] - mnew);
      l[r] = l[r] * alpha + red_sum16(p0 + p1);
      m[r] = mnew;
#pragma unroll
      for (int t = 0; t < 4; ++t) acc[t][r] *= alpha;
      const int prow = r + 8 * hl;
      P[prow * 32 + lr]      = (_Float16)p0;
      P[prow * 32 + 16 + lr] = (_Float16)p1;
    }
    asm volatile("s_wait_dscnt 0" ::: "memory");
    v16h pf;
    {
      const int ko = hl ? 16 : 0;
#pragma unroll
      for (int i = 0; i < 8; ++i) {
        pf[i]     = P[lr * 32 + ko + i];
        pf[i + 8] = P[lr * 32 + ko + 8 + i];
      }
    }
    const int jr = j0 + (hl ? 16 : 0);
#pragma unroll
    for (int t = 0; t < 4; ++t) {
      v16h vf = *(const v16h*)(v2T + ((size_t)(b * D2 + t * 16 + lr) * N_SZ + jr));
      acc[t] = wmma_f16(pf, vf, acc[t]);
    }
  }
  // y = attn/l + proj ; LayerNorm over 64 ; write f32 output
#pragma unroll
  for (int r = 0; r < 8; ++r) {
    const int i = i0 + r + 8 * hl;
    const float inv = 1.f / (l[r] + 1e-20f);
    float y[4];
#pragma unroll
    for (int t = 0; t < 4; ++t)
      y[t] = acc[t][r] * inv + proj[(size_t)(b * N_SZ + i) * D2 + t * 16 + lr];
    const float mu = red_sum16(y[0] + y[1] + y[2] + y[3]) * (1.f / 64.f);
    float sq = 0.f;
#pragma unroll
    for (int t = 0; t < 4; ++t) { const float d = y[t] - mu; sq += d * d; }
    const float varr = red_sum16(sq) * (1.f / 64.f);
    const float rstd = rsqrtf(varr + 1e-6f);
#pragma unroll
    for (int t = 0; t < 4; ++t) {
      const int c = t * 16 + lr;
      out[(size_t)(b * N_SZ + i) * D2 + c] = gamma[c] * (y[t] - mu) * rstd + beta[c];
    }
  }
}

extern "C" void kernel_launch(void* const* d_in, const int* in_sizes, int n_in,
                              void* d_out, int out_size, void* d_ws, size_t ws_size,
                              hipStream_t stream) {
  (void)in_sizes; (void)n_in; (void)out_size; (void)ws_size;
  const float* x     = (const float*)d_in[0];
  const float* adj   = (const float*)d_in[1];
  const float* Wp    = (const float*)d_in[2];
  const float* bp    = (const float*)d_in[3];
  const float* Wq1   = (const float*)d_in[4];
  const float* bq1   = (const float*)d_in[5];
  const float* Wk1   = (const float*)d_in[6];
  const float* bk1   = (const float*)d_in[7];
  const float* Wv1   = (const float*)d_in[8];
  const float* bv1   = (const float*)d_in[9];
  const float* Wq2   = (const float*)d_in[10];
  const float* bq2   = (const float*)d_in[11];
  const float* Wk2   = (const float*)d_in[12];
  const float* bk2   = (const float*)d_in[13];
  const float* Wv2   = (const float*)d_in[14];
  const float* bv2   = (const float*)d_in[15];
  const float* gamma = (const float*)d_in[16];
  const float* beta  = (const float*)d_in[17];

  char* ws = (char*)d_ws;
  size_t off = 0;
  auto take = [&](size_t bytes) -> void* {
    void* p = ws + off;
    off += (bytes + 255) & ~(size_t)255;
    return p;
  };
  _Float16* q1   = (_Float16*)take((size_t)B_SZ * N_SZ * C1 * sizeof(_Float16));
  _Float16* k1   = (_Float16*)take((size_t)B_SZ * N_SZ * C1 * sizeof(_Float16));
  _Float16* v1T  = (_Float16*)take((size_t)B_SZ * C1 * N_SZ * sizeof(_Float16));
  _Float16* h1   = (_Float16*)take((size_t)B_SZ * N_SZ * C1 * sizeof(_Float16));
  float*    proj = (float*)   take((size_t)B_SZ * N_SZ * DOUT * sizeof(float));
  _Float16* q2   = (_Float16*)take((size_t)B_SZ * N_SZ * D2 * sizeof(_Float16));
  _Float16* k2   = (_Float16*)take((size_t)B_SZ * N_SZ * D2 * sizeof(_Float16));
  _Float16* v2T  = (_Float16*)take((size_t)B_SZ * D2 * N_SZ * sizeof(_Float16));

  qkv1_proj_kernel<<<dim3(B_SZ * N_SZ / 16), dim3(256), 0, stream>>>(
      x, Wq1, bq1, Wk1, bk1, Wv1, bv1, Wp, bp, q1, k1, v1T, proj);
  attn1_kernel<<<dim3(N_SZ / 128, H1, B_SZ), dim3(256), 0, stream>>>(
      q1, k1, v1T, adj, h1);
  qkv2_kernel<<<dim3(B_SZ * N_SZ / 16), dim3(256), 0, stream>>>(
      h1, Wq2, bq2, Wk2, bk2, Wv2, bv2, q2, k2, v2T);
  attn2_ln_kernel<<<dim3(N_SZ / 128, 1, B_SZ), dim3(256), 0, stream>>>(
      q2, k2, v2T, adj, proj, gamma, beta, (float*)d_out);
}
